// InteractionModule_73486890434697
// MI455X (gfx1250) — matrix operand order repaired and checked by reading, not compiled, not run on hardware
//
#include <hip/hip_runtime.h>
#include <hip/hip_bf16.h>

#define E_EDGES 100000
#define N_NODES 5000
#define INV_SQRT3 0.57735026918962576451f
#define AVG_INV (1.0f/20.0f)
#define R_MAX 5.0f

typedef __bf16 bf16_t;
typedef __attribute__((ext_vector_type(16))) __bf16 v16bf;
typedef __attribute__((ext_vector_type(8)))  float  v8f;

enum { EPI_NONE = 0, EPI_SILU = 1, EPI_CUT = 2, EPI_RESID = 3 };

__device__ __forceinline__ float siluf(float x) { return x / (1.0f + __expf(-x)); }

// ---------------------------------------------------------------------------
// Tensor Data Mover support (gfx1250): 2-D tile load Global->LDS, bf16 data,
// tile 32(k) x rows(<=64), with LDS padding of 2 DWORDs every 16 DWORDs so the
// stored stride is 36 bf16 (matches AS_STRIDE/BS_STRIDE bank-conflict padding).
// This toolchain exposes the 6-argument builtin:
//   (uint32x4 g0, int32x8 g1, int32x4 g2, int32x4 g3, int32x8 gx, i32 cpol)
// ---------------------------------------------------------------------------
#if defined(__HIP_DEVICE_COMPILE__) && __has_builtin(__builtin_amdgcn_tensor_load_to_lds)
#define HAVE_TDM 1
typedef __attribute__((ext_vector_type(4))) unsigned int u32x4;
typedef __attribute__((ext_vector_type(8))) int i32x8;
typedef __attribute__((ext_vector_type(4))) int i32x4;

__device__ __forceinline__ void tdm_load_tile(unsigned int lds_byte_off,
                                              const void* gtile,
                                              int rows, int ld_elems)
{
    unsigned long long ga = (unsigned long long)(size_t)gtile;
    u32x4 g0;
    g0[0] = 1u;                                   // count=1, user descriptor
    g0[1] = lds_byte_off;                         // lds_addr
    g0[2] = (unsigned int)ga;                     // global_addr[31:0]
    g0[3] = (unsigned int)((ga >> 32) & 0x01FFFFFFu) | (2u << 30);  // [56:32] | type=2
    i32x8 g1;
    // data_size=2B, pad_enable, pad_interval=16 DWORDs, pad_amount=2 DWORDs
    g1[0] = (int)((1u << 16) | (1u << 20) | (3u << 22) | (1u << 25));
    g1[1] = (int)(32u << 16);                     // tensor_dim0 = 32
    g1[2] = (int)(((unsigned)rows & 0xFFFFu) << 16); // tensor_dim1 = rows (OOB rows -> 0)
    g1[3] = (int)(32u << 16);                     // tile_dim0 = 32
    g1[4] = 64;                                   // tile_dim1 = 64
    g1[5] = ld_elems;                             // tensor_dim0_stride (elements)
    g1[6] = 0; g1[7] = 0;
    i32x4 gz4 = {0, 0, 0, 0};
    i32x8 gz8 = {0, 0, 0, 0, 0, 0, 0, 0};
    __builtin_amdgcn_tensor_load_to_lds(g0, g1, gz4, gz4, gz8, 0);
}
#endif

__device__ __forceinline__ void wait_tensor0()
{
#if defined(HAVE_TDM)
#if __has_builtin(__builtin_amdgcn_s_wait_tensorcnt)
    __builtin_amdgcn_s_wait_tensorcnt(0);
#else
    asm volatile("s_wait_tensorcnt 0x0" ::: "memory");
#endif
#endif
}

// ---------------------------------------------------------------------------
// WMMA GEMM: C[M x N] = A[M x K] @ B[K x N], B pre-transposed bf16 [N][ldb].
// Block = 128 threads (4 waves). Block tile 64(M) x 64(N); wave w owns rows
// [16w,16w+16) across four 16-wide N subtiles. K stepped by 32 via LDS with
// double buffering (one barrier per K-step). B tile (and bf16 A tile) staged
// by the Tensor Data Mover when available; f32 A converted by vector pipes.
// A/B fragments use contiguous k = 16*(lane/16) + j (same order for both); f32 C/D per ISA.
// ---------------------------------------------------------------------------
#define AS_STRIDE 36
#define BS_STRIDE 36

template <int ABF16>
__global__ void __launch_bounds__(128)
gemm_wmma(const void* __restrict__ Avoid, int lda,
          const bf16_t* __restrict__ Bt, int ldb,
          float* __restrict__ C, int ldc,
          int M, int K, int epi,
          const float* __restrict__ cut,
          const float* __restrict__ rsrc, int ldr,
          const float* __restrict__ resp)
{
    __shared__ bf16_t As[2][64 * AS_STRIDE];
    __shared__ bf16_t Bs[2][64 * BS_STRIDE];

    const int tid  = threadIdx.x;
    const int lane = tid & 31;
    const int wv   = tid >> 5;
    const int m0   = blockIdx.x * 64;
    const int n0   = blockIdx.y * 64;

    v8f acc[4];
    #pragma unroll
    for (int t = 0; t < 4; ++t)
        for (int i = 0; i < 8; ++i) acc[t][i] = 0.0f;

    const int arow = tid >> 1;        // 0..63 staging row
    const int acol = (tid & 1) * 16;  // 0 or 16
    const int grow = m0 + arow;
    const int nk   = K >> 5;

    auto stageAB = [&](int ktx, int b) {
        const int ks = ktx * 32;
#ifdef HAVE_TDM
        if (wv == 0) {
            tdm_load_tile((unsigned int)(size_t)&Bs[b][0],
                          (const void*)(Bt + (size_t)n0 * ldb + ks), 64, ldb);
            if (ABF16) {
                int rows = M - m0; if (rows > 64) rows = 64;
                tdm_load_tile((unsigned int)(size_t)&As[b][0],
                              (const void*)((const bf16_t*)Avoid + (size_t)m0 * lda + ks),
                              rows, lda);
            }
        }
        if (!ABF16) {
            bf16_t* d = &As[b][arow * AS_STRIDE + acol];
            if (grow < M) {
                const float* a = (const float*)Avoid + (size_t)grow * lda + ks + acol;
                #pragma unroll
                for (int j = 0; j < 16; ++j) d[j] = (bf16_t)a[j];
                if (ktx + 1 < nk) __builtin_prefetch((const void*)(a + 32), 0, 1);
            } else {
                #pragma unroll
                for (int j = 0; j < 16; ++j) d[j] = (bf16_t)0.0f;
            }
        }
#else
        {   // ---- synchronous A staging ----
            bf16_t* d = &As[b][arow * AS_STRIDE + acol];
            if (grow < M) {
                if (ABF16) {
                    const bf16_t* a = (const bf16_t*)Avoid + (size_t)grow * lda + ks + acol;
                    #pragma unroll
                    for (int j = 0; j < 16; ++j) d[j] = a[j];
                    if (ktx + 1 < nk) __builtin_prefetch((const void*)(a + 32), 0, 1);
                } else {
                    const float* a = (const float*)Avoid + (size_t)grow * lda + ks + acol;
                    #pragma unroll
                    for (int j = 0; j < 16; ++j) d[j] = (bf16_t)a[j];
                    if (ktx + 1 < nk) __builtin_prefetch((const void*)(a + 32), 0, 1);
                }
            } else {
                #pragma unroll
                for (int j = 0; j < 16; ++j) d[j] = (bf16_t)0.0f;
            }
        }
        {   // ---- synchronous B staging ----
            const bf16_t* bsrc = Bt + (size_t)(n0 + arow) * ldb + ks + acol;
            bf16_t* d = &Bs[b][arow * BS_STRIDE + acol];
            #pragma unroll
            for (int j = 0; j < 16; ++j) d[j] = bsrc[j];
        }
#endif
    };

    int buf = 0;
    stageAB(0, 0);
    wait_tensor0();
    __syncthreads();

    for (int kt = 0; kt < nk; ++kt) {
        const int nxt = buf ^ 1;
        if (kt + 1 < nk) stageAB(kt + 1, nxt);

        // ---- build fragments & WMMA from buffer `buf` ----
        // A (16-bit 16x32): lanes0-15 hold K 0..15, lanes16-31 K 16..31
        const int rl = wv * 16 + (lane & 15);
        const int ah = (lane >> 4) << 4;   // 0 or 16
        const bf16_t* ap = &As[buf][rl * AS_STRIDE];
        v16bf af;
        #pragma unroll
        for (int j = 0; j < 16; ++j) { af[j] = ap[ah + j]; }

        // B (16-bit 32x16): lanes0-15 K=0..15, lanes16-31 K=16..31, col = lane&15
        const int kbb = (lane >> 4) << 4;  // 0 or 16
        #pragma unroll
        for (int t = 0; t < 4; ++t) {
            const bf16_t* bp = &Bs[buf][(t * 16 + (lane & 15)) * BS_STRIDE + kbb];
            v16bf bfr;
            #pragma unroll
            for (int j = 0; j < 16; ++j) bfr[j] = bp[j];
            acc[t] = __builtin_amdgcn_wmma_f32_16x16x32_bf16(
                false, af, false, bfr, (short)0, acc[t], false, false);
        }

        wait_tensor0();
        __syncthreads();
        buf = nxt;
    }

    // ---- epilogue ----
    float sa = 1.0f, sb = 0.0f;
    if (epi == EPI_RESID) {
        float a = 1.0f / (1.0f + __expf(-resp[1]));
        sa = sqrtf(1.0f - a);
        sb = sqrtf(a);
    }
    // C/D layout: VGPR j -> M = j + (lane>=16 ? 8 : 0), N = lane & 15
    const int mbase = wv * 16 + ((lane >> 4) << 3);
    const int nl    = lane & 15;
    #pragma unroll
    for (int t = 0; t < 4; ++t) {
        const int gc = n0 + t * 16 + nl;
        #pragma unroll
        for (int j = 0; j < 8; ++j) {
            const int gm = m0 + mbase + j;
            if (gm < M) {
                float v = acc[t][j];
                if (epi == EPI_SILU)      v = siluf(v);
                else if (epi == EPI_CUT)  v = v * cut[gm];
                else if (epi == EPI_RESID)
                    v = sa * rsrc[(size_t)gm * ldr + gc] + sb * cut[gm] * v;
                C[(size_t)gm * ldc + gc] = v;
            }
        }
    }
}

// ---------------------------------------------------------------------------
// Weight prep: fp32 [K][N] -> bf16 transposed [N][Kpad] (zero pad k >= K)
// ---------------------------------------------------------------------------
__global__ void wprep_t(const float* __restrict__ src, bf16_t* __restrict__ dst,
                        int K, int N, int Kpad)
{
    int idx = blockIdx.x * blockDim.x + threadIdx.x;
    if (idx >= N * Kpad) return;
    int n = idx / Kpad, k = idx - n * Kpad;
    dst[idx] = (k < K) ? (bf16_t)src[(size_t)k * N + n] : (bf16_t)0.0f;
}

// lin1 weights [2,2,64,64] (i,o,u,v) -> bf16 [(o*64+v)][(i*64+u)]
__global__ void wprep_lin1(const float* __restrict__ src, bf16_t* __restrict__ dst)
{
    int t = blockIdx.x * blockDim.x + threadIdx.x;
    if (t >= 16384) return;
    int v = t & 63, u = (t >> 6) & 63, o = (t >> 12) & 1, i = (t >> 13) & 1;
    dst[(o * 64 + v) * 128 + (i * 64 + u)] = (bf16_t)src[t];
}

// ---------------------------------------------------------------------------
// Edge gather: X0[e][160] = bf16 concat(node[c], node[n], radial, pad0); cut[e]
// ---------------------------------------------------------------------------
__global__ void __launch_bounds__(160)
edge_gather(const float* __restrict__ nodes, const float* __restrict__ radial,
            const float* __restrict__ elen, const int* __restrict__ eidx,
            bf16_t* __restrict__ X0, float* __restrict__ cut)
{
    int e = blockIdx.x, k = threadIdx.x;
    int c = eidx[e], n = eidx[E_EDGES + e];
    float v;
    if (k < 64)        v = nodes[(size_t)c * 64 + k];
    else if (k < 128)  v = nodes[(size_t)n * 64 + (k - 64)];
    else if (k < 136)  v = radial[(size_t)e * 8 + (k - 128)];
    else               v = 0.0f;
    X0[(size_t)e * 160 + k] = (bf16_t)v;
    if (k == 0) {
        float x = elen[e] * (1.0f / R_MAX);
        float x2 = x * x, x6 = x2 * x2 * x2, x7 = x6 * x, x8 = x7 * x;
        float o = 1.0f - 28.0f * x6 + 48.0f * x7 - 21.0f * x8;
        cut[e] = (x < 1.0f) ? o : 0.0f;
    }
}

__global__ void zero_f32(float* __restrict__ p, int n)
{
    int i = blockIdx.x * blockDim.x + threadIdx.x;
    if (i < n) p[i] = 0.0f;
}

// ---------------------------------------------------------------------------
// Env weighting + gating + segment-sum scatter (atomic).
// wfeat_off >= 0: features from w_all; else gated from fs_in/fv_in planes.
// ---------------------------------------------------------------------------
__global__ void __launch_bounds__(128)
env_scatter(const float* __restrict__ wall, int ldw, int gate_off, int wfeat_off,
            const float* __restrict__ sh, const int* __restrict__ center,
            const float* __restrict__ fs_in, const float* __restrict__ fv_in,
            float* __restrict__ fs_out, float* __restrict__ fv_out,
            float* __restrict__ Ns, float* __restrict__ Nv)
{
    int e = blockIdx.x * 2 + (threadIdx.x >> 6);
    int u = threadIdx.x & 63;
    if (e >= E_EDGES) return;
    const float* w = wall + (size_t)e * ldw;
    float s0 = sh[4 * e + 0], s1 = sh[4 * e + 1], s2 = sh[4 * e + 2], s3 = sh[4 * e + 3];
    float g = w[gate_off + u];
    float fs, f0, f1, f2;
    const size_t PE = (size_t)E_EDGES * 64;
    size_t p = (size_t)e * 64 + u;
    if (wfeat_off >= 0) {
        float a = w[wfeat_off + 2 * u], b = w[wfeat_off + 2 * u + 1];
        fs = a * s0; f0 = b * s1; f1 = b * s2; f2 = b * s3;
    } else {
        fs = fs_in[p]; f0 = fv_in[p]; f1 = fv_in[PE + p]; f2 = fv_in[2 * PE + p];
    }
    fs *= g; f0 *= g; f1 *= g; f2 *= g;
    fs_out[p] = fs;
    fv_out[p] = f0; fv_out[PE + p] = f1; fv_out[2 * PE + p] = f2;

    float we0 = w[2 * u], we1 = w[2 * u + 1];
    int cn = center[e];
    size_t q = (size_t)cn * 64 + u;
    const size_t PN = (size_t)N_NODES * 64;
    atomicAdd(&Ns[q], we0 * s0);
    atomicAdd(&Nv[q], we1 * s1);
    atomicAdd(&Nv[PN + q], we1 * s2);
    atomicAdd(&Nv[2 * PN + q], we1 * s3);
}

// Node-level 64x64 linears (envlin): ns2 = (Ns/AVG)@W[0], nv2 = (Nv/AVG)@W[1]
__global__ void __launch_bounds__(64)
node_lin(const float* __restrict__ Ns, const float* __restrict__ Nv,
         const float* __restrict__ W, float* __restrict__ ns2, float* __restrict__ nv2)
{
    int n = blockIdx.x, v = threadIdx.x;
    const size_t PN = (size_t)N_NODES * 64;
    float as = 0.f, a0 = 0.f, a1 = 0.f, a2 = 0.f;
    for (int u = 0; u < 64; ++u) {
        float ws = W[u * 64 + v];
        float wv = W[4096 + u * 64 + v];
        size_t q = (size_t)n * 64 + u;
        as += Ns[q] * ws;
        a0 += Nv[q] * wv;
        a1 += Nv[PN + q] * wv;
        a2 += Nv[2 * PN + q] * wv;
    }
    size_t o = (size_t)n * 64 + v;
    ns2[o] = as * AVG_INV;
    nv2[o] = a0 * AVG_INV;
    nv2[PN + o] = a1 * AVG_INV;
    nv2[2 * PN + o] = a2 * AVG_INV;
}

// uuu tensor product: scalars -> lat[:,256:384], vectors -> TV planes [3][E][128]
__global__ void __launch_bounds__(128)
tp_uuu(const float* __restrict__ fs, const float* __restrict__ fv,
       const float* __restrict__ ns2, const float* __restrict__ nv2,
       const int* __restrict__ center,
       float* __restrict__ lat_scalars, int ldl, float* __restrict__ TV)
{
    int e = blockIdx.x * 2 + (threadIdx.x >> 6);
    int u = threadIdx.x & 63;
    if (e >= E_EDGES) return;
    int cn = center[e];
    const size_t PE = (size_t)E_EDGES * 64;
    const size_t PN = (size_t)N_NODES * 64;
    const size_t PT = (size_t)E_EDGES * 128;
    size_t pe = (size_t)e * 64 + u;
    size_t pn = (size_t)cn * 64 + u;
    float s  = fs[pe];
    float v0 = fv[pe], v1 = fv[PE + pe], v2 = fv[2 * PE + pe];
    float ns = ns2[pn];
    float n0 = nv2[pn], n1 = nv2[PN + pn], n2 = nv2[2 * PN + pn];
    float* L = lat_scalars + (size_t)e * ldl;
    L[u]      = s * ns;
    L[64 + u] = (v0 * n0 + v1 * n1 + v2 * n2) * INV_SQRT3;
    float* T = TV + (size_t)e * 128;
    T[u]            = s * n0 * INV_SQRT3;  T[64 + u]            = v0 * ns * INV_SQRT3;
    T[PT + u]       = s * n1 * INV_SQRT3;  T[PT + 64 + u]       = v1 * ns * INV_SQRT3;
    T[2 * PT + u]   = s * n2 * INV_SQRT3;  T[2 * PT + 64 + u]   = v2 * ns * INV_SQRT3;
}

// out[e] = concat(FS1[e]*FIN[e] (128), FV1 interleaved (o*64+v)*3+c (384))
__global__ void __launch_bounds__(128)
finalize(const float* __restrict__ FS1, const float* __restrict__ FV1,
         const float* __restrict__ FIN, float* __restrict__ out)
{
    int e = blockIdx.x, t = threadIdx.x;
    const size_t PF = (size_t)E_EDGES * 128;
    float* o = out + (size_t)e * 512;
    o[t] = FS1[(size_t)e * 128 + t] * FIN[(size_t)e * 128 + t];
    const float* fv = FV1 + (size_t)e * 128 + t;
    o[128 + 3 * t + 0] = fv[0];
    o[128 + 3 * t + 1] = fv[PF];
    o[128 + 3 * t + 2] = fv[2 * PF];
}

// ---------------------------------------------------------------------------
extern "C" void kernel_launch(void* const* d_in, const int* in_sizes, int n_in,
                              void* d_out, int out_size, void* d_ws, size_t ws_size,
                              hipStream_t stream)
{
    (void)in_sizes; (void)n_in; (void)out_size; (void)ws_size;
    const float* node_attrs = (const float*)d_in[0];
    const float* radial     = (const float*)d_in[1];
    const float* sh         = (const float*)d_in[2];
    const float* elen       = (const float*)d_in[3];
    const int*   eidx       = (const int*)d_in[4];
    const float* tb_w0      = (const float*)d_in[5];
    const float* tb_w1      = (const float*)d_in[6];
    const float* tb_w2      = (const float*)d_in[7];
    const float* lat1_w0    = (const float*)d_in[8];
    const float* lat1_w1    = (const float*)d_in[9];
    const float* env0_w     = (const float*)d_in[10];
    const float* env1_w     = (const float*)d_in[11];
    const float* envlin0    = (const float*)d_in[12];
    const float* envlin1    = (const float*)d_in[13];
    const float* lin0_sw    = (const float*)d_in[14];
    const float* lin0_vw    = (const float*)d_in[15];
    const float* lin1_sw    = (const float*)d_in[16];
    const float* lin1_vw    = (const float*)d_in[17];
    const float* fin_w0     = (const float*)d_in[18];
    const float* fin_w1     = (const float*)d_in[19];
    const float* res_p      = (const float*)d_in[20];
    float* out = (float*)d_out;

    const size_t E = E_EDGES, N = N_NODES;
    char* base = (char*)d_ws; size_t off = 0;
    auto alloc = [&](size_t bytes) -> void* {
        void* p = base + off; off = (off + bytes + 255) & ~(size_t)255; return p;
    };

    bf16_t* Wtb0t   = (bf16_t*)alloc(256 * 160 * 2);
    bf16_t* Wtb1t   = (bf16_t*)alloc(256 * 256 * 2);
    bf16_t* Wtb2t   = (bf16_t*)alloc(256 * 256 * 2);
    bf16_t* Wenv0t  = (bf16_t*)alloc(320 * 256 * 2);
    bf16_t* Wlat10t = (bf16_t*)alloc(256 * 384 * 2);
    bf16_t* Wlat11t = (bf16_t*)alloc(256 * 256 * 2);
    bf16_t* Wenv1t  = (bf16_t*)alloc(192 * 256 * 2);
    bf16_t* Wlin0st = (bf16_t*)alloc(64 * 128 * 2);
    bf16_t* Wlin0vt = (bf16_t*)alloc(64 * 128 * 2);
    bf16_t* Wlin1st = (bf16_t*)alloc(128 * 128 * 2);
    bf16_t* Wlin1vt = (bf16_t*)alloc(128 * 128 * 2);
    bf16_t* Wfin0t  = (bf16_t*)alloc(256 * 384 * 2);
    bf16_t* Wfin1t  = (bf16_t*)alloc(128 * 256 * 2);

    bf16_t* X0  = (bf16_t*)alloc(E * 160 * 2);
    float* cut  = (float*)alloc(E * 4);
    float* H1   = (float*)alloc(E * 256 * 4);
    float* H2   = (float*)alloc(E * 256 * 4);
    float* LAT1 = (float*)alloc(E * 384 * 4);
    float* LAT2 = (float*)alloc(E * 384 * 4);
    float* WALL = (float*)alloc(E * 320 * 4);      // reused (192-wide) in pass 2
    float* FSg  = (float*)alloc(E * 64 * 4);
    float* FVg  = (float*)alloc(3 * E * 64 * 4);
    float* Ns   = (float*)alloc(N * 64 * 4);       // contiguous with Nv
    float* Nv   = (float*)alloc(3 * N * 64 * 4);
    float* ns2  = (float*)alloc(N * 64 * 4);
    float* nv2  = (float*)alloc(3 * N * 64 * 4);
    float* TV   = (float*)alloc(3 * E * 128 * 4);  // reused in pass 2
    float* FS0  = (float*)alloc(E * 64 * 4);
    float* FV0  = (float*)alloc(3 * E * 64 * 4);
    float* FS1  = (float*)alloc(E * 128 * 4);
    float* FV1  = (float*)alloc(3 * E * 128 * 4);
    float* FIN  = (float*)alloc(E * 128 * 4);

    auto TP = [&](const float* s, bf16_t* d, int K, int Nn, int Kpad) {
        int cnt = Nn * Kpad;
        wprep_t<<<(cnt + 255) / 256, 256, 0, stream>>>(s, d, K, Nn, Kpad);
    };
    TP(tb_w0,   Wtb0t,   136, 256, 160);
    TP(tb_w1,   Wtb1t,   256, 256, 256);
    TP(tb_w2,   Wtb2t,   256, 256, 256);
    TP(env0_w,  Wenv0t,  256, 320, 256);
    TP(lat1_w0, Wlat10t, 384, 256, 384);
    TP(lat1_w1, Wlat11t, 256, 256, 256);
    TP(env1_w,  Wenv1t,  256, 192, 256);
    TP(lin0_sw, Wlin0st, 128, 64, 128);
    TP(lin0_vw, Wlin0vt, 128, 64, 128);
    TP(fin_w0,  Wfin0t,  384, 256, 384);
    TP(fin_w1,  Wfin1t,  256, 128, 256);
    wprep_lin1<<<64, 256, 0, stream>>>(lin1_sw, Wlin1st);
    wprep_lin1<<<64, 256, 0, stream>>>(lin1_vw, Wlin1vt);

    edge_gather<<<(int)E, 160, 0, stream>>>(node_attrs, radial, elen, eidx, X0, cut);

    auto GEMM = [&](int abf, const void* A, int lda, const bf16_t* Bt, int ldb,
                    float* C, int ldc, int M, int K, int Nn, int epi,
                    const float* cutp, const float* rs, int ldr) {
        dim3 g((M + 63) / 64, Nn / 64), b(128);
        if (abf) gemm_wmma<1><<<g, b, 0, stream>>>(A, lda, Bt, ldb, C, ldc, M, K, epi, cutp, rs, ldr, res_p);
        else     gemm_wmma<0><<<g, b, 0, stream>>>(A, lda, Bt, ldb, C, ldc, M, K, epi, cutp, rs, ldr, res_p);
    };

    const int Ei = (int)E;
    const size_t PE64 = E * 64, PE128 = E * 128;

    // two-body MLP -> latents (in LAT1[:, :256])
    GEMM(1, X0, 160, Wtb0t, 160, H1, 256, Ei, 160, 256, EPI_SILU, nullptr, nullptr, 0);
    GEMM(0, H1, 256, Wtb1t, 256, H2, 256, Ei, 256, 256, EPI_SILU, nullptr, nullptr, 0);
    GEMM(0, H2, 256, Wtb2t, 256, LAT1, 384, Ei, 256, 256, EPI_CUT, cut, nullptr, 0);
    // env0 projection
    GEMM(0, LAT1, 384, Wenv0t, 256, WALL, 320, Ei, 256, 320, EPI_NONE, nullptr, nullptr, 0);

    zero_f32<<<((int)(4 * N * 64) + 255) / 256, 256, 0, stream>>>(Ns, (int)(4 * N * 64));
    env_scatter<<<Ei / 2, 128, 0, stream>>>(WALL, 320, 128, 192, sh, eidx,
                                            nullptr, nullptr, FSg, FVg, Ns, Nv);
    node_lin<<<(int)N, 64, 0, stream>>>(Ns, Nv, envlin0, ns2, nv2);
    tp_uuu<<<Ei / 2, 128, 0, stream>>>(FSg, FVg, ns2, nv2, eidx, LAT1 + 256, 384, TV);

    // lin0 (scalars K=128 -> 64; vectors per component)
    GEMM(0, LAT1 + 256, 384, Wlin0st, 128, FS0, 64, Ei, 128, 64, EPI_NONE, nullptr, nullptr, 0);
    for (int c = 0; c < 3; ++c)
        GEMM(0, TV + (size_t)c * PE128, 128, Wlin0vt, 128, FV0 + (size_t)c * PE64, 64,
             Ei, 128, 64, EPI_NONE, nullptr, nullptr, 0);

    // lat1 MLP + residual mix -> LAT2[:, :256]
    GEMM(0, LAT1, 384, Wlat10t, 384, H1, 256, Ei, 384, 256, EPI_SILU, nullptr, nullptr, 0);
    GEMM(0, H1, 256, Wlat11t, 256, LAT2, 384, Ei, 256, 256, EPI_RESID, cut, LAT1, 384);

    // env1 projection
    GEMM(0, LAT2, 384, Wenv1t, 256, WALL, 192, Ei, 256, 192, EPI_NONE, nullptr, nullptr, 0);

    zero_f32<<<((int)(4 * N * 64) + 255) / 256, 256, 0, stream>>>(Ns, (int)(4 * N * 64));
    env_scatter<<<Ei / 2, 128, 0, stream>>>(WALL, 192, 128, -1, sh, eidx,
                                            FS0, FV0, FSg, FVg, Ns, Nv);
    node_lin<<<(int)N, 64, 0, stream>>>(Ns, Nv, envlin1, ns2, nv2);
    tp_uuu<<<Ei / 2, 128, 0, stream>>>(FSg, FVg, ns2, nv2, eidx, LAT2 + 256, 384, TV);

    // lin1 (K=128 -> 128)
    GEMM(0, LAT2 + 256, 384, Wlin1st, 128, FS1, 128, Ei, 128, 128, EPI_NONE, nullptr, nullptr, 0);
    for (int c = 0; c < 3; ++c)
        GEMM(0, TV + (size_t)c * PE128, 128, Wlin1vt, 128, FV1 + (size_t)c * PE128, 128,
             Ei, 128, 128, EPI_NONE, nullptr, nullptr, 0);

    // final MLP
    GEMM(0, LAT2, 384, Wfin0t, 384, H1, 256, Ei, 384, 256, EPI_SILU, nullptr, nullptr, 0);
    GEMM(0, H1, 256, Wfin1t, 256, FIN, 128, Ei, 256, 128, EPI_NONE, nullptr, nullptr, 0);

    finalize<<<Ei, 128, 0, stream>>>(FS1, FV1, FIN, out);
}
